// SparseAttention_77927886619054
// MI455X (gfx1250) — matrix operand order repaired and checked by reading, not compiled, not run on hardware
//
#include <hip/hip_runtime.h>
#include <hip/hip_bf16.h>

// ---------------------------------------------------------------------------
// NSA forward for gfx1250 (MI455X).  Compile-only target.
//
// Dims (from reference): B=1 S=1024 D=1024 H=16 HK=4 DH=64 G=4 CBS=32 SBS=32
// NSEL=8 WIN=64 NMEM=1 DIM_INNER=1024 DIM_KV=256 CDIM=2048 NCB=32 NFB=32 NWB=16
//
// ~10.5 GFLOP total, 85% in GEMMs -> v_wmma_f32_16x16x32_f16 (f16 in, f32 acc).
// Working set (~60MB) fits in 192MB L2; HBM streaming cost ~2.5us at 23.3TB/s,
// so the design goal is matrix-pipe utilization + latency hiding:
//  - double-buffered LDS tiles, A staged with global_load_async_to_lds_b128
//    (ASYNCcnt) overlapped with WMMA compute on the other buffer
//  - all fragments loaded before issuing 4 back-to-back v_wmma ops
// ---------------------------------------------------------------------------

#define S_   1024
#define D_   1024
#define H_   16
#define HK_  4
#define DH_  64
#define G_   4
#define CBS_ 32
#define SBS_ 32
#define NSEL_ 8
#define WIN_ 64
#define NCB_ 32
#define NWB_ 16
#define CDIM_ 2048
#define QKVW_ 1536          // DIM_INNER + 2*DIM_KV
#define SCALE_ 0.125f       // DH^-0.5
#define NEGF_ (-3.402823466e38f)

typedef _Float16 v16h __attribute__((ext_vector_type(16)));
typedef float    v8f  __attribute__((ext_vector_type(8)));

// ---------------------------------------------------------------------------
// f32 -> f16 elementwise convert (weights)
// ---------------------------------------------------------------------------
__global__ void f32_to_f16_kernel(const float* __restrict__ src,
                                  _Float16* __restrict__ dst, int n) {
  int i = blockIdx.x * blockDim.x + threadIdx.x;
  if (i < n) dst[i] = (_Float16)src[i];
}

// ---------------------------------------------------------------------------
// RMSNorm: one block per row (D=1024), output f16
// ---------------------------------------------------------------------------
__global__ __launch_bounds__(256) void rmsnorm_kernel(
    const float* __restrict__ inp, const float* __restrict__ w,
    _Float16* __restrict__ x16) {
  int s = blockIdx.x, tid = threadIdx.x;
  __shared__ float red[256];
  const float* row = inp + (size_t)s * D_;
  float ss = 0.f;
  for (int d = tid; d < D_; d += 256) { float v = row[d]; ss += v * v; }
  red[tid] = ss;
  for (int st = 128; st; st >>= 1) {
    __syncthreads();
    if (tid < st) red[tid] += red[tid + st];
  }
  __syncthreads();
  float inv = rsqrtf(red[0] * (1.0f / D_) + 1e-6f);
  for (int d = tid; d < D_; d += 256)
    x16[(size_t)s * D_ + d] = (_Float16)(row[d] * inv * w[d]);
}

// ---------------------------------------------------------------------------
// WMMA GEMM: C[M,N] = A[M,K] * B[K,N] (+bias, relu), A/B f16 row-major.
// Block tile 128x64, 8 wave32 waves (each wave: 16 rows x 64 cols = 4 subtiles)
// K-step 32, double-buffered LDS.  A tile staged with async global->LDS
// (pure f16 copy, ASYNCcnt-tracked); B tile transposed to [N][K] through
// registers so each lane's fragment is two contiguous ds_load_b128 per the
// CDNA5 16-bit layouts:
//   A 16x32: lane L -> row L&15, halves 0..7 = K (L>=16?8:0)+0..7,
//            halves 8..15 = K 16+(L>=16?8:0)+0..7
//   B 32x16: lane L -> col L&15, same K order as A
//   C 16x16 f32: vgpr r -> M = r + (L>=16?8:0), N = L&15
// M % 128 == 0, N % 64 == 0, K % 32 == 0 for every call in this model.
// ---------------------------------------------------------------------------
__global__ __launch_bounds__(256) void wmma_gemm_kernel(
    const _Float16* __restrict__ A, const _Float16* __restrict__ B,
    const float* __restrict__ bias, float* __restrict__ C32,
    _Float16* __restrict__ C16, int M, int N, int K, int relu) {
  __shared__ __align__(16) _Float16 As[2][128 * 32];
  __shared__ __align__(16) _Float16 Bt[2][64 * 32];

  int tid = threadIdx.x, wid = tid >> 5, lane = tid & 31;
  int n0 = blockIdx.x * 64, m0 = blockIdx.y * 128;
  int la = lane & 15, hi = lane >> 4;

  v8f acc[4];
  for (int j = 0; j < 4; ++j)
    for (int r = 0; r < 8; ++r) acc[j][r] = 0.f;

  int arow = tid >> 1, acol = (tid & 1) * 16;  // 256 thr * 16 f16 = 128x32
  int bk = tid >> 3, bn = (tid & 7) * 8;       // 256 thr * 8  f16 = 32x64

  // Stage one K-tile into LDS buffer `buf`:
  //  - A: two async global->LDS b128 copies per thread (32 bytes)
  //  - B: b128 global load + 8x ds_store_b16 transpose into Bt[n][k]
  auto stage = [&](int buf, int k0) {
    unsigned gbyte =
        (unsigned)(((size_t)(m0 + arow) * K + k0 + acol) * sizeof(_Float16));
    unsigned lds0 = (unsigned)(size_t)&As[buf][arow * 32 + acol];
    asm volatile(
        "global_load_async_to_lds_b128 %0, %1, %2\n\t"
        "global_load_async_to_lds_b128 %3, %4, %2"
        :
        : "v"(lds0), "v"(gbyte), "s"(A), "v"(lds0 + 16u), "v"(gbyte + 16u)
        : "memory");
    union { uint4 q; _Float16 hf[8]; } bu;
    bu.q = *reinterpret_cast<const uint4*>(B + (size_t)(k0 + bk) * N + n0 + bn);
    for (int jj = 0; jj < 8; ++jj) Bt[buf][(bn + jj) * 32 + bk] = bu.hf[jj];
  };

  stage(0, 0);
  asm volatile("s_wait_asynccnt 0x0" ::: "memory");
  __syncthreads();

  int buf = 0;
  for (int k0 = 0; k0 < K; k0 += 32) {
    int kn = k0 + 32;
    if (kn < K) stage(buf ^ 1, kn);  // overlap next-tile staging with compute

    const _Float16* as_ = As[buf];
    const _Float16* bt_ = Bt[buf];
    union Frag { v16h v; uint4 q[2]; } af, bf[4];
    int am = wid * 16 + la;
    af.q[0] = *reinterpret_cast<const uint4*>(&as_[am * 32 + hi * 8]);
    af.q[1] = *reinterpret_cast<const uint4*>(&as_[am * 32 + 16 + hi * 8]);
    for (int j = 0; j < 4; ++j) {
      int bn2 = j * 16 + la;
      bf[j].q[0] = *reinterpret_cast<const uint4*>(&bt_[bn2 * 32 + hi * 8]);
      bf[j].q[1] =
          *reinterpret_cast<const uint4*>(&bt_[bn2 * 32 + 16 + hi * 8]);
    }
    // back-to-back WMMA issue on the XDL pipe
    for (int j = 0; j < 4; ++j)
      acc[j] = __builtin_amdgcn_wmma_f32_16x16x32_f16(
          false, af.v, false, bf[j].v, (short)0, acc[j], false, false);

    asm volatile("s_wait_asynccnt 0x0" ::: "memory");
    __syncthreads();
    buf ^= 1;
  }

  for (int j = 0; j < 4; ++j) {
    int n = n0 + j * 16 + la;
    float bv = bias ? bias[n] : 0.f;
    for (int r = 0; r < 8; ++r) {
      int m = m0 + wid * 16 + r + hi * 8;
      float v = acc[j][r] + bv;
      if (relu) v = fmaxf(v, 0.f);
      size_t off = (size_t)m * N + n;
      if (C32) C32[off] = v;
      if (C16) C16[off] = (_Float16)v;
    }
  }
}

// ---------------------------------------------------------------------------
// Interleaved RoPE (rotary-embedding-torch convention), theta=10000.
// src[s*srcStride + srcOff + h*64 + d] -> dst[s*dstStride + h*64 + d]
// ---------------------------------------------------------------------------
__global__ void rope_kernel(const float* __restrict__ src, int srcStride,
                            int srcOff, float* __restrict__ dst, int dstStride,
                            int nh, int total) {
  int idx = blockIdx.x * blockDim.x + threadIdx.x;
  if (idx >= total) return;
  int i = idx & 31;                 // pair index (DH/2 = 32)
  int h = (idx >> 5) % nh;
  int s = idx / (32 * nh);
  const float* sp = src + (size_t)s * srcStride + srcOff + h * DH_ + 2 * i;
  float x0 = sp[0], x1 = sp[1];
  float fr = powf(10000.0f, -(float)(2 * i) * (1.0f / (float)DH_));
  float ang = (float)s * fr;
  float c = cosf(ang), sn = sinf(ang);
  float* dp = dst + (size_t)s * dstStride + h * DH_ + 2 * i;
  dp[0] = x0 * c - x1 * sn;
  dp[1] = x1 * c + x0 * sn;
}

// ---------------------------------------------------------------------------
// Build compress-branch blocks: kb/vb[(hk*NCB+c)*CDIM + i*DH + d] (f16)
//   = k/v[s=c*32+i, hk, d] + k_pos/v_pos[hk, i, d]
// ---------------------------------------------------------------------------
__global__ void build_blocks_kernel(const float* __restrict__ qkv,
                                    const float* __restrict__ kpos,
                                    const float* __restrict__ vpos,
                                    _Float16* __restrict__ kb,
                                    _Float16* __restrict__ vb) {
  int idx = blockIdx.x * blockDim.x + threadIdx.x;  // HK*NCB*CBS*DH = 262144
  if (idx >= HK_ * NCB_ * CBS_ * DH_) return;
  int d = idx & 63;
  int i = (idx >> 6) & 31;
  int c = (idx >> 11) & 31;
  int hk = idx >> 16;
  int s = c * CBS_ + i;
  size_t q = (size_t)s * QKVW_ + hk * DH_ + d;
  size_t p = (size_t)hk * CBS_ * DH_ + i * DH_ + d;
  size_t o = (size_t)(hk * NCB_ + c) * CDIM_ + i * DH_ + d;
  kb[o] = (_Float16)(qkv[q + 1024] + kpos[p]);
  vb[o] = (_Float16)(qkv[q + 1280] + vpos[p]);
}

// ---------------------------------------------------------------------------
// Coarse compressed attention: one thread per (h, s), 33 keys (mem + 32 blocks)
// q is un-roped (from qkv); c_out layout (S, H, DH).
// ---------------------------------------------------------------------------
__global__ __launch_bounds__(256) void coarse_attn_kernel(
    const float* __restrict__ qkv, const float* __restrict__ ck,
    const float* __restrict__ cv, const float* __restrict__ mem_kv,
    float* __restrict__ c_out) {
  int tid = blockIdx.x * blockDim.x + threadIdx.x;  // H*S
  if (tid >= H_ * S_) return;
  int h = tid >> 10, s = tid & (S_ - 1), hk = h >> 2;
  const float* qp = qkv + (size_t)s * QKVW_ + h * DH_;
  const float* mk = mem_kv + hk * DH_;                 // mem_kv[0,hk,0,:]
  const float* mv = mem_kv + HK_ * DH_ + hk * DH_;     // mem_kv[1,hk,0,:]

  float sc[33];
  float mx;
  {
    float acc = 0.f;
    for (int d = 0; d < DH_; ++d) acc += qp[d] * mk[d];
    sc[0] = acc * SCALE_;            // mem key: always causal (-1 < s)
    mx = sc[0];
  }
  for (int c = 0; c < NCB_; ++c) {
    const float* kp = ck + (size_t)(hk * NCB_ + c) * DH_;
    float acc = 0.f;
    for (int d = 0; d < DH_; ++d) acc += qp[d] * kp[d];
    float v = (s >= (c + 1) * CBS_) ? acc * SCALE_ : NEGF_;
    sc[c + 1] = v;
    mx = fmaxf(mx, v);
  }
  float sum = 0.f;
  for (int j = 0; j < 33; ++j) {
    float e = __expf(sc[j] - mx);
    sc[j] = e;
    sum += e;
  }
  float inv = 1.f / sum;
  float* op = c_out + (size_t)s * (H_ * DH_) + h * DH_;
  for (int d = 0; d < DH_; ++d) {
    float acc = sc[0] * mv[d];
    for (int c = 0; c < NCB_; ++c)
      acc += sc[c + 1] * cv[(size_t)(hk * NCB_ + c) * DH_ + d];
    op[d] = acc * inv;
  }
}

// ---------------------------------------------------------------------------
// Importance (mean over grouped q of coarse scores) -> softmax -> top-8.
// One thread per (hk, s).  selidx/selok: [HK, S, 9] (slot 8 = own block).
// ---------------------------------------------------------------------------
__global__ __launch_bounds__(256) void topk_kernel(
    const float* __restrict__ qkv, const float* __restrict__ ck,
    int* __restrict__ selidx, int* __restrict__ selok) {
  int tid = blockIdx.x * blockDim.x + threadIdx.x;  // HK*S
  if (tid >= HK_ * S_) return;
  int hk = tid >> 10, s = tid & (S_ - 1);

  float sc[33];
  sc[0] = -1e3f;
  float mx = -1e3f;
  for (int c = 0; c < NCB_; ++c) {
    const float* kp = ck + (size_t)(hk * NCB_ + c) * DH_;
    float acc = 0.f;
    for (int g = 0; g < G_; ++g) {
      const float* qp = qkv + (size_t)s * QKVW_ + (hk * G_ + g) * DH_;
      for (int d = 0; d < DH_; ++d) acc += qp[d] * kp[d];
    }
    float v = (s >= (c + 1) * CBS_) ? acc * (0.25f * SCALE_) : NEGF_;
    sc[c + 1] = v;
    mx = fmaxf(mx, v);
  }
  float sum = 0.f;
  for (int j = 0; j < 33; ++j) sum += __expf(sc[j] - mx);
  float inv = 1.f / sum;
  float pr[NCB_];
  for (int c = 0; c < NCB_; ++c) pr[c] = __expf(sc[c + 1] - mx) * inv;

  size_t base = (size_t)(hk * S_ + s) * 9;
  unsigned used = 0u;
  for (int t = 0; t < NSEL_; ++t) {
    int best = 0;
    float bv = -1.f;
    for (int c = 0; c < NCB_; ++c)
      if (!((used >> c) & 1u) && pr[c] > bv) { bv = pr[c]; best = c; }
    used |= 1u << best;
    selidx[base + t] = best;
    selok[base + t] = (bv > 1e-10f) ? 1 : 0;
  }
  selidx[base + 8] = s >> 5;  // own block
  selok[base + 8] = 1;
}

// ---------------------------------------------------------------------------
// Fine selected-block attention.  One block per (s, hk), 4 waves = 4 grouped
// heads.  9 blocks x 32 keys staged as f16 in LDS; per-lane 9 keys; wave
// shfl-xor softmax reduction; values read from global (L2-resident).
// ---------------------------------------------------------------------------
__global__ __launch_bounds__(128) void fine_attn_kernel(
    const float* __restrict__ rq, const float* __restrict__ rk,
    const float* __restrict__ qkv, const int* __restrict__ selidx,
    const int* __restrict__ selok, float* __restrict__ f_out) {
  int s = blockIdx.x, hk = blockIdx.y;
  int tid = threadIdx.x, g = tid >> 5, lane = tid & 31;
  int h = hk * G_ + g;

  __shared__ int sidx[9];
  __shared__ int sok[9];
  __shared__ float qs[G_ * DH_];
  __shared__ __align__(16) _Float16 fks[288 * DH_];
  __shared__ float ps[G_ * 288];

  if (tid < 9) {
    size_t base = (size_t)(hk * S_ + s) * 9;
    sidx[tid] = selidx[base + tid];
    sok[tid] = selok[base + tid];
  }
  for (int e = tid; e < G_ * DH_; e += 128) {
    int gg = e >> 6, d = e & 63;
    qs[e] = rq[(size_t)s * (H_ * DH_) + (hk * G_ + gg) * DH_ + d];
  }
  __syncthreads();
  for (int e = tid; e < 288 * DH_; e += 128) {
    int r = e >> 6, d = e & 63;
    int skey = sidx[r >> 5] * SBS_ + (r & 31);
    fks[e] = (_Float16)rk[(size_t)skey * (HK_ * DH_) + hk * DH_ + d];
  }
  __syncthreads();

  const float* q = qs + g * DH_;
  int srow = s & 31;
  float loc[9];
  float mx = NEGF_;
  for (int t = 0; t < 9; ++t) {
    int j = t * 32 + lane;  // key-in-block == lane
    const _Float16* kp = fks + (size_t)j * DH_;
    float acc = 0.f;
    for (int d = 0; d < DH_; ++d) acc += q[d] * (float)kp[d];
    acc *= SCALE_;
    bool ok = (t < NSEL_) ? (sok[t] != 0) : (lane <= srow);
    loc[t] = ok ? acc : NEGF_;
    mx = fmaxf(mx, loc[t]);
  }
  for (int m = 16; m; m >>= 1) mx = fmaxf(mx, __shfl_xor(mx, m, 32));
  float lsum = 0.f;
  for (int t = 0; t < 9; ++t) {
    float e = __expf(loc[t] - mx);
    ps[g * 288 + t * 32 + lane] = e;
    lsum += e;
  }
  for (int m = 16; m; m >>= 1) lsum += __shfl_xor(lsum, m, 32);
  float inv = 1.f / lsum;
  __syncthreads();

  for (int d2 = lane; d2 < DH_; d2 += 32) {
    float acc = 0.f;
    for (int t = 0; t < 9; ++t) {
      int sb = sidx[t] * SBS_;
      for (int i2 = 0; i2 < SBS_; ++i2) {
        float vv = qkv[(size_t)(sb + i2) * QKVW_ + 1280 + hk * DH_ + d2];
        acc += ps[g * 288 + t * 32 + i2] * vv;
      }
    }
    f_out[(size_t)s * (H_ * DH_) + h * DH_ + d2] = acc * inv;
  }
}

// ---------------------------------------------------------------------------
// Sliding-window attention.  Block per (w, h): 64 queries, 128 keys
// (prev + current window) staged as f16 in LDS; 8 waves, each wave owns
// 8 queries; exact window mask diff in [0, WIN] and key pos >= 0.
// ---------------------------------------------------------------------------
__global__ __launch_bounds__(256) void slide_attn_kernel(
    const float* __restrict__ rq, const float* __restrict__ rk,
    const float* __restrict__ qkv, float* __restrict__ s_out) {
  int w = blockIdx.x, h = blockIdx.y, hk = h >> 2;
  int tid = threadIdx.x, wid = tid >> 5, lane = tid & 31;

  __shared__ __align__(16) _Float16 k2s[128 * DH_];
  __shared__ __align__(16) _Float16 v2s[128 * DH_];
  __shared__ float qsl[WIN_ * DH_];
  __shared__ float ps2[8 * 128];

  for (int e = tid; e < 128 * DH_; e += 256) {
    int jj = e >> 6, d = e & 63;
    int kpos = (w - 1) * WIN_ + jj;
    float kv = 0.f, vv = 0.f;
    if (kpos >= 0) {
      kv = rk[(size_t)kpos * (HK_ * DH_) + hk * DH_ + d];
      vv = qkv[(size_t)kpos * QKVW_ + 1280 + hk * DH_ + d];
    }
    k2s[e] = (_Float16)kv;
    v2s[e] = (_Float16)vv;
  }
  for (int e = tid; e < WIN_ * DH_; e += 256) {
    int i = e >> 6, d = e & 63;
    qsl[e] = rq[(size_t)(w * WIN_ + i) * (H_ * DH_) + h * DH_ + d];
  }
  __syncthreads();

  for (int i = wid; i < WIN_; i += 8) {
    const float* q = qsl + i * DH_;
    float loc[4];
    float mx = NEGF_;
    for (int t = 0; t < 4; ++t) {
      int jj = t * 32 + lane;
      const _Float16* kp = k2s + (size_t)jj * DH_;
      float acc = 0.f;
      for (int d = 0; d < DH_; ++d) acc += q[d] * (float)kp[d];
      acc *= SCALE_;
      bool ok = (jj >= i) && (jj <= i + WIN_) && ((w * WIN_ + jj - WIN_) >= 0);
      loc[t] = ok ? acc : NEGF_;
      mx = fmaxf(mx, loc[t]);
    }
    for (int m = 16; m; m >>= 1) mx = fmaxf(mx, __shfl_xor(mx, m, 32));
    float lsum = 0.f;
    for (int t = 0; t < 4; ++t) {
      float e = __expf(loc[t] - mx);
      ps2[wid * 128 + t * 32 + lane] = e;
      lsum += e;
    }
    for (int m = 16; m; m >>= 1) lsum += __shfl_xor(lsum, m, 32);
    float inv = 1.f / lsum;
    for (int d2 = lane; d2 < DH_; d2 += 32) {
      float acc = 0.f;
      for (int jj = 0; jj < 128; ++jj)
        acc += ps2[wid * 128 + jj] * (float)v2s[jj * DH_ + d2];
      s_out[(size_t)(w * WIN_ + i) * (H_ * DH_) + h * DH_ + d2] = acc * inv;
    }
  }
}

// ---------------------------------------------------------------------------
// Gate projection: sigmoid(x @ comb_w + comb_b), one thread per (s, col)
// ---------------------------------------------------------------------------
__global__ void gates_kernel(const _Float16* __restrict__ x16,
                             const float* __restrict__ comb_w,
                             const float* __restrict__ comb_b,
                             float* __restrict__ gates) {
  int idx = blockIdx.x * blockDim.x + threadIdx.x;  // S * 48
  if (idx >= S_ * 48) return;
  int s = idx / 48, c = idx % 48;
  float acc = comb_b[c];
  const _Float16* xp = x16 + (size_t)s * D_;
  for (int k = 0; k < D_; ++k) acc += (float)xp[k] * comb_w[(size_t)k * 48 + c];
  gates[idx] = 1.f / (1.f + __expf(-acc));
}

// ---------------------------------------------------------------------------
// Gated combine of the 3 branches -> f16 for final Wo GEMM
// ---------------------------------------------------------------------------
__global__ void combine_kernel(const float* __restrict__ c_out,
                               const float* __restrict__ f_out,
                               const float* __restrict__ s_out,
                               const float* __restrict__ gates,
                               _Float16* __restrict__ comb16) {
  int idx = blockIdx.x * blockDim.x + threadIdx.x;  // S * H * DH
  if (idx >= S_ * H_ * DH_) return;
  int s = idx >> 10, h = (idx >> 6) & 15;
  const float* g = gates + (size_t)s * 48 + h * 3;
  float v = c_out[idx] * g[0] + f_out[idx] * g[1] + s_out[idx] * g[2];
  comb16[idx] = (_Float16)v;
}

// ---------------------------------------------------------------------------
// Launch
// ---------------------------------------------------------------------------
extern "C" void kernel_launch(void* const* d_in, const int* in_sizes, int n_in,
                              void* d_out, int out_size, void* d_ws,
                              size_t ws_size, hipStream_t stream) {
  (void)in_sizes; (void)n_in; (void)out_size; (void)ws_size;
  const float* inp    = (const float*)d_in[0];
  const float* norm_w = (const float*)d_in[1];
  const float* Wqkv   = (const float*)d_in[2];
  const float* k_pos  = (const float*)d_in[3];
  const float* v_pos  = (const float*)d_in[4];
  const float* mem_kv = (const float*)d_in[5];
  const float* kc_w1  = (const float*)d_in[6];
  const float* kc_b1  = (const float*)d_in[7];
  const float* kc_w2  = (const float*)d_in[8];
  const float* kc_b2  = (const float*)d_in[9];
  const float* vc_w1  = (const float*)d_in[10];
  const float* vc_b1  = (const float*)d_in[11];
  const float* vc_w2  = (const float*)d_in[12];
  const float* vc_b2  = (const float*)d_in[13];
  const float* comb_w = (const float*)d_in[14];
  const float* comb_b = (const float*)d_in[15];
  const float* Wo     = (const float*)d_in[16];
  float* out = (float*)d_out;

  // --- workspace carve (all fully overwritten every call; ~51 MB) ---
  size_t off = 0;
  char* wsb = (char*)d_ws;
  auto alloc = [&](size_t nbytes) -> char* {
    char* p = wsb + off;
    off = (off + nbytes + 255) & ~(size_t)255;
    return p;
  };
  _Float16* x16     = (_Float16*)alloc((size_t)S_ * D_ * 2);
  _Float16* Wqkv16  = (_Float16*)alloc((size_t)D_ * QKVW_ * 2);
  float*    qkvf    = (float*)   alloc((size_t)S_ * QKVW_ * 4);
  float*    rqf     = (float*)   alloc((size_t)S_ * H_ * DH_ * 4);
  float*    rkf     = (float*)   alloc((size_t)S_ * HK_ * DH_ * 4);
  _Float16* kb16    = (_Float16*)alloc((size_t)HK_ * NCB_ * CDIM_ * 2);
  _Float16* vb16    = (_Float16*)alloc((size_t)HK_ * NCB_ * CDIM_ * 2);
  _Float16* kcw1_16 = (_Float16*)alloc((size_t)CDIM_ * CDIM_ * 2);
  _Float16* vcw1_16 = (_Float16*)alloc((size_t)CDIM_ * CDIM_ * 2);
  _Float16* kcw2_16 = (_Float16*)alloc((size_t)CDIM_ * DH_ * 2);
  _Float16* vcw2_16 = (_Float16*)alloc((size_t)CDIM_ * DH_ * 2);
  _Float16* hk16    = (_Float16*)alloc((size_t)HK_ * NCB_ * CDIM_ * 2);
  _Float16* hv16    = (_Float16*)alloc((size_t)HK_ * NCB_ * CDIM_ * 2);
  float*    ckf     = (float*)   alloc((size_t)HK_ * NCB_ * DH_ * 4);
  float*    cvf     = (float*)   alloc((size_t)HK_ * NCB_ * DH_ * 4);
  float*    c_outf  = (float*)   alloc((size_t)S_ * H_ * DH_ * 4);
  int*      selidx  = (int*)     alloc((size_t)HK_ * S_ * 9 * 4);
  int*      selok   = (int*)     alloc((size_t)HK_ * S_ * 9 * 4);
  float*    f_outf  = (float*)   alloc((size_t)S_ * H_ * DH_ * 4);
  float*    s_outf  = (float*)   alloc((size_t)S_ * H_ * DH_ * 4);
  float*    gatesf  = (float*)   alloc((size_t)S_ * 48 * 4);
  _Float16* comb16  = (_Float16*)alloc((size_t)S_ * H_ * DH_ * 2);
  _Float16* Wo16    = (_Float16*)alloc((size_t)D_ * D_ * 2);

  auto cvt = [&](const float* s, _Float16* d, int n) {
    f32_to_f16_kernel<<<(n + 255) / 256, 256, 0, stream>>>(s, d, n);
  };
  cvt(Wqkv, Wqkv16, D_ * QKVW_);
  cvt(kc_w1, kcw1_16, CDIM_ * CDIM_);
  cvt(vc_w1, vcw1_16, CDIM_ * CDIM_);
  cvt(kc_w2, kcw2_16, CDIM_ * DH_);
  cvt(vc_w2, vcw2_16, CDIM_ * DH_);
  cvt(Wo, Wo16, D_ * D_);

  // RMSNorm -> x (f16)
  rmsnorm_kernel<<<S_, 256, 0, stream>>>(inp, norm_w, x16);

  // QKV projection: (1024 x 1024) @ (1024 x 1536)
  wmma_gemm_kernel<<<dim3(QKVW_ / 64, S_ / 128), 256, 0, stream>>>(
      x16, Wqkv16, nullptr, qkvf, nullptr, S_, QKVW_, D_, 0);

  // RoPE for q (H heads) and k (HK heads)
  rope_kernel<<<(S_ * H_ * 32 + 255) / 256, 256, 0, stream>>>(
      qkvf, QKVW_, 0, rqf, H_ * DH_, H_, S_ * H_ * 32);
  rope_kernel<<<(S_ * HK_ * 32 + 255) / 256, 256, 0, stream>>>(
      qkvf, QKVW_, 1024, rkf, HK_ * DH_, HK_, S_ * HK_ * 32);

  // compress-branch block build + two-layer MLP compressors (WMMA)
  build_blocks_kernel<<<(HK_ * NCB_ * CBS_ * DH_ + 255) / 256, 256, 0,
                        stream>>>(qkvf, k_pos, v_pos, kb16, vb16);
  wmma_gemm_kernel<<<dim3(CDIM_ / 64, 1), 256, 0, stream>>>(
      kb16, kcw1_16, kc_b1, nullptr, hk16, HK_ * NCB_, CDIM_, CDIM_, 1);
  wmma_gemm_kernel<<<dim3(CDIM_ / 64, 1), 256, 0, stream>>>(
      vb16, vcw1_16, vc_b1, nullptr, hv16, HK_ * NCB_, CDIM_, CDIM_, 1);
  wmma_gemm_kernel<<<dim3(1, 1), 256, 0, stream>>>(
      hk16, kcw2_16, kc_b2, ckf, nullptr, HK_ * NCB_, DH_, CDIM_, 0);
  wmma_gemm_kernel<<<dim3(1, 1), 256, 0, stream>>>(
      hv16, vcw2_16, vc_b2, cvf, nullptr, HK_ * NCB_, DH_, CDIM_, 0);

  // branch 1: coarse attention; then importance top-k selection
  coarse_attn_kernel<<<(H_ * S_) / 256, 256, 0, stream>>>(qkvf, ckf, cvf,
                                                          mem_kv, c_outf);
  topk_kernel<<<(HK_ * S_) / 256, 256, 0, stream>>>(qkvf, ckf, selidx, selok);

  // branch 2: fine selected-block attention
  fine_attn_kernel<<<dim3(S_, HK_), 128, 0, stream>>>(rqf, rkf, qkvf, selidx,
                                                      selok, f_outf);
  // branch 3: sliding window
  slide_attn_kernel<<<dim3(NWB_, H_), 256, 0, stream>>>(rqf, rkf, qkvf, s_outf);

  // gates + combine + output projection
  gates_kernel<<<(S_ * 48 + 255) / 256, 256, 0, stream>>>(x16, comb_w, comb_b,
                                                          gatesf);
  combine_kernel<<<(S_ * H_ * DH_) / 256, 256, 0, stream>>>(
      c_outf, f_outf, s_outf, gatesf, comb16);
  wmma_gemm_kernel<<<dim3(D_ / 64, S_ / 128), 256, 0, stream>>>(
      comb16, Wo16, nullptr, out, nullptr, S_, D_, H_ * DH_, 0);
}
